// BiMamba_block_60309930771129
// MI455X (gfx1250) — matrix operand order repaired and checked by reading, not compiled, not run on hardware
//
#include <hip/hip_runtime.h>
#include <hip/hip_bf16.h>
#include <math.h>

// ---------------------------------------------------------------------------
// BiMamba block for MI455X (gfx1250, wave32, WMMA).
//  - GEMMs: bf16 WMMA (v_wmma_f32_16x16x32_bf16), fp32 accumulate.
//    f32 -> bf16 conversion happens while staging LDS tiles with unguarded
//    float4 loads (K is always a multiple of 64; dt GEMM inputs are
//    zero-padded 48 -> 64 by a prep kernel).
//  - conv / softplus / selective scan / gating / LayerNorm: fp32.
//  - Workspace: ~564 MB fp32 scratch.
// ---------------------------------------------------------------------------

#define D_MODEL 768
#define D_INNER 1536
#define D_STATE 16
#define DT_RANK 48
#define BATCH   8
#define SEQ     2048
#define NTOK    (BATCH * SEQ)   // 16384

typedef __attribute__((ext_vector_type(16))) __bf16 v16bf;
typedef __attribute__((ext_vector_type(4)))  __bf16 bf16x4;
typedef __attribute__((ext_vector_type(8)))  float  v8f;
typedef __attribute__((ext_vector_type(4)))  float  f32x4;

// ---------------------------------------------------------------------------
// GEMM:  C[M,N] (+)= A[M,K] * B[N,K]^T   (f32 in global, bf16 WMMA math)
// Requirements: K % 64 == 0, lda/ldb % 4 == 0, 16B-aligned base pointers.
// Block tile 128(M) x 128(N), staged K-step 64 (two 32-deep WMMA substeps).
// 8 waves: 4 (M) x 2 (N); each wave owns 32x64 via 2x4 16x16 accumulators.
// ---------------------------------------------------------------------------
#define BLK_M 128
#define BLK_N 128
#define BLK_K 64
#define LDS_K 72   // padded row stride (bf16 elems) to dodge bank conflicts

__global__ __launch_bounds__(256)
void gemm_bf16_wmma(const float* __restrict__ A, int lda,
                    const float* __restrict__ B, int ldb,
                    float* __restrict__ C, int ldc,
                    int M, int N, int K, int beta)
{
    __shared__ __bf16 As[BLK_M * LDS_K];
    __shared__ __bf16 Bs[BLK_N * LDS_K];

    const int tid  = threadIdx.x;
    const int lane = tid & 31;
    const int wave = tid >> 5;
    const int wm   = wave & 3;          // 4 wave-rows  -> 32 M each
    const int wn   = wave >> 2;         // 2 wave-cols  -> 64 N each
    const int block_m = blockIdx.y * BLK_M;
    const int block_n = blockIdx.x * BLK_N;

    v8f zero = {};
    v8f acc[2][4] = {{zero, zero, zero, zero}, {zero, zero, zero, zero}};

    for (int k0 = 0; k0 < K; k0 += BLK_K) {
        // ---- prefetch next K-tile into near caches (global_prefetch_b8).
        // Prefetch is speculative: safe without bounds guards (ISA drops on
        // translation failure), so no exec-mask churn in the hot loop.
        if (k0 + BLK_K < K) {
            int prow = tid >> 1;                 // 0..127
            int pcol = k0 + BLK_K + (tid & 1) * 32;
            __builtin_prefetch(A + (size_t)(block_m + prow) * lda + pcol, 0, 3);
            __builtin_prefetch(B + (size_t)(block_n + prow) * ldb + pcol, 0, 3);
        }

        // ---- stage A tile: 128x64 f32 -> bf16, float4 loads, b64 LDS stores
        #pragma unroll
        for (int i = 0; i < 8; ++i) {
            int idx = tid + i * 256;        // 0..2047
            int row = idx >> 4;             // 0..127
            int c4  = (idx & 15) * 4;       // 0..60
            int gm  = block_m + row;
            f32x4 v = {};
            if (gm < M)
                v = *reinterpret_cast<const f32x4*>(A + (size_t)gm * lda + k0 + c4);
            bf16x4 h;
            h[0] = (__bf16)v[0]; h[1] = (__bf16)v[1];
            h[2] = (__bf16)v[2]; h[3] = (__bf16)v[3];
            *reinterpret_cast<bf16x4*>(&As[row * LDS_K + c4]) = h;
        }
        // ---- stage B tile: 128x64, same scheme (row-granular N guard)
        #pragma unroll
        for (int i = 0; i < 8; ++i) {
            int idx = tid + i * 256;
            int row = idx >> 4;
            int c4  = (idx & 15) * 4;
            int gn  = block_n + row;
            f32x4 v = {};
            if (gn < N)
                v = *reinterpret_cast<const f32x4*>(B + (size_t)gn * ldb + k0 + c4);
            bf16x4 h;
            h[0] = (__bf16)v[0]; h[1] = (__bf16)v[1];
            h[2] = (__bf16)v[2]; h[3] = (__bf16)v[3];
            *reinterpret_cast<bf16x4*>(&Bs[row * LDS_K + c4]) = h;
        }
        __syncthreads();

        // ---- two 32-deep WMMA substeps over the staged 64-wide tile
        #pragma unroll
        for (int ks = 0; ks < 2; ++ks) {
            v16bf afrag[2], bfrag[4];
            const int kbA = (lane >> 4) * 16;  // A 16x32 lane layout
            #pragma unroll
            for (int mt = 0; mt < 2; ++mt) {
                const __bf16* ap = &As[(wm * 32 + mt * 16 + (lane & 15)) * LDS_K + ks * 32];
                #pragma unroll
                for (int j = 0; j < 8; ++j) {
                    afrag[mt][j]     = ap[kbA + j];
                    afrag[mt][j + 8] = ap[kbA + 8 + j];
                }
            }
            const int kbB = (lane >> 4) * 16;  // B 32x16 lane layout
            #pragma unroll
            for (int nt = 0; nt < 4; ++nt) {
                const __bf16* bp = &Bs[(wn * 64 + nt * 16 + (lane & 15)) * LDS_K + ks * 32];
                #pragma unroll
                for (int j = 0; j < 16; ++j)
                    bfrag[nt][j] = bp[kbB + j];
            }
            #pragma unroll
            for (int mt = 0; mt < 2; ++mt)
                #pragma unroll
                for (int nt = 0; nt < 4; ++nt)
                    acc[mt][nt] = __builtin_amdgcn_wmma_f32_16x16x32_bf16(
                        false, afrag[mt], false, bfrag[nt],
                        (short)0, acc[mt][nt], false, false);
        }
        __syncthreads();
    }

    // ---- epilogue (C 16x16 f32 layout: vgpr r -> M=r / M=8+r, lane&15 -> N)
    const int halfsel = (lane >> 4) * 8;
    #pragma unroll
    for (int mt = 0; mt < 2; ++mt) {
        #pragma unroll
        for (int nt = 0; nt < 4; ++nt) {
            int gn = block_n + wn * 64 + nt * 16 + (lane & 15);
            #pragma unroll
            for (int r = 0; r < 8; ++r) {
                int gm = block_m + wm * 32 + mt * 16 + halfsel + r;
                if (gm < M && gn < N) {
                    size_t off = (size_t)gm * ldc + gn;
                    float v = acc[mt][nt][r];
                    C[off] = beta ? (C[off] + v) : v;
                }
            }
        }
    }
}

// ---------------------------------------------------------------------------
// Zero-pad a 48-wide (strided) f32 matrix into a dense 64-wide one.
// ---------------------------------------------------------------------------
__global__ __launch_bounds__(256)
void pad48to64_kernel(const float* __restrict__ src, int srcld,
                      float* __restrict__ dst, int rows)
{
    size_t idx = (size_t)blockIdx.x * 256 + threadIdx.x;
    if (idx >= (size_t)rows * 64) return;
    int c = (int)(idx & 63);
    size_t r = idx >> 6;
    dst[idx] = (c < DT_RANK) ? src[r * srcld + c] : 0.0f;
}

// ---------------------------------------------------------------------------
// Depthwise causal conv (kernel 4) + SiLU.  reverse=1 -> conv over flipped time.
// ---------------------------------------------------------------------------
__global__ __launch_bounds__(256)
void conv_silu_kernel(const float* __restrict__ xz,
                      const float* __restrict__ conv_w,
                      const float* __restrict__ conv_b,
                      float* __restrict__ uc, int reverse)
{
    size_t idx = (size_t)blockIdx.x * 256 + threadIdx.x;
    if (idx >= (size_t)NTOK * D_INNER) return;
    int d = (int)(idx % D_INNER);
    int l = (int)((idx / D_INNER) % SEQ);
    int b = (int)(idx / ((size_t)D_INNER * SEQ));

    const float* u = xz + (size_t)b * SEQ * (2 * D_INNER) + d;
    float acc = conv_b[d];
    #pragma unroll
    for (int k = 0; k < 4; ++k) {
        int t = reverse ? (l + 3 - k) : (l - 3 + k);
        if (t >= 0 && t < SEQ)
            acc += conv_w[d * 4 + k] * u[(size_t)t * (2 * D_INNER)];
    }
    uc[idx] = acc / (1.0f + __expf(-acc));   // SiLU
}

// ---------------------------------------------------------------------------
// dt = softplus(dt_raw + dt_b)
// ---------------------------------------------------------------------------
__global__ __launch_bounds__(256)
void softplus_kernel(float* __restrict__ dt, const float* __restrict__ dt_b)
{
    size_t idx = (size_t)blockIdx.x * 256 + threadIdx.x;
    if (idx >= (size_t)NTOK * D_INNER) return;
    int d = (int)(idx % D_INNER);
    float x = dt[idx] + dt_b[d];
    dt[idx] = (x > 20.0f) ? x : logf(1.0f + __expf(x));
}

// ---------------------------------------------------------------------------
// Selective scan. One thread per (b, d) channel, 16 states in registers.
// B_t/C_t (shared across channels) staged 16 timesteps at a time via LDS.
// ---------------------------------------------------------------------------
__global__ __launch_bounds__(256)
void scan_kernel(const float* __restrict__ dtbuf,
                 const float* __restrict__ ucbuf,
                 const float* __restrict__ xdbl,
                 const float* __restrict__ A_log,
                 float* __restrict__ ybuf, int reverse)
{
    __shared__ float sBC[16][32];      // [step in chunk][0..15 = B, 16..31 = C]
    const int tid = threadIdx.x;
    const int d   = blockIdx.x * 256 + tid;   // < D_INNER (6 blocks * 256)
    const int b   = blockIdx.y;

    float Arow[D_STATE];
    #pragma unroll
    for (int n = 0; n < D_STATE; ++n)
        Arow[n] = -__expf(A_log[(size_t)d * D_STATE + n]);

    float h[D_STATE];
    #pragma unroll
    for (int n = 0; n < D_STATE; ++n) h[n] = 0.0f;

    const size_t base80 = (size_t)b * SEQ * (DT_RANK + 2 * D_STATE);
    const size_t baseDI = (size_t)b * SEQ * D_INNER;

    for (int c = 0; c < SEQ / 16; ++c) {
        __syncthreads();
        for (int i = tid; i < 512; i += 256) {
            int j = i >> 5;            // step within chunk
            int v = i & 31;            // 0..31 (B then C)
            int t = c * 16 + j;
            if (reverse) t = SEQ - 1 - t;
            sBC[j][v] = xdbl[base80 + (size_t)t * (DT_RANK + 2 * D_STATE) + DT_RANK + v];
        }
        __syncthreads();

        for (int j = 0; j < 16; ++j) {
            int t = c * 16 + j;
            if (reverse) t = SEQ - 1 - t;
            size_t off = baseDI + (size_t)t * D_INNER + d;
            float dt = dtbuf[off];
            float u  = ucbuf[off];
            float du = dt * u;
            float y  = 0.0f;
            #pragma unroll
            for (int n = 0; n < D_STATE; ++n) {
                float dA = __expf(dt * Arow[n]);       // TRANS pipe, co-executes w/ VALU
                h[n] = dA * h[n] + du * sBC[j][n];
                y   += h[n] * sBC[j][16 + n];
            }
            ybuf[off] = y;
        }
    }
}

// ---------------------------------------------------------------------------
// y = (y + uc*D_skip) * silu(z)    (z is second half of xz rows)
// ---------------------------------------------------------------------------
__global__ __launch_bounds__(256)
void gate_kernel(float* __restrict__ y, const float* __restrict__ uc,
                 const float* __restrict__ xz, const float* __restrict__ D_skip)
{
    size_t idx = (size_t)blockIdx.x * 256 + threadIdx.x;
    if (idx >= (size_t)NTOK * D_INNER) return;
    int d = (int)(idx % D_INNER);
    size_t row = idx / D_INNER;
    float z = xz[row * (2 * D_INNER) + D_INNER + d];
    float g = z / (1.0f + __expf(-z));
    y[idx] = (y[idx] + uc[idx] * D_skip[d]) * g;
}

// ---------------------------------------------------------------------------
// out = LayerNorm(out_acc + residual) * g + b     (one block per token row)
// ---------------------------------------------------------------------------
__global__ __launch_bounds__(256)
void ln_kernel(const float* __restrict__ acc, const float* __restrict__ x,
               const float* __restrict__ g, const float* __restrict__ bta,
               float* __restrict__ out)
{
    __shared__ float rs[256], rq[256];
    const int row = blockIdx.x;
    const int tid = threadIdx.x;
    const size_t base = (size_t)row * D_MODEL;

    float local[3];
    float s = 0.0f, s2 = 0.0f;
    #pragma unroll
    for (int i = 0; i < 3; ++i) {
        int c = tid + i * 256;
        float v = acc[base + c] + x[base + c];
        local[i] = v; s += v; s2 += v * v;
    }
    rs[tid] = s; rq[tid] = s2;
    __syncthreads();
    for (int o = 128; o > 0; o >>= 1) {
        if (tid < o) { rs[tid] += rs[tid + o]; rq[tid] += rq[tid + o]; }
        __syncthreads();
    }
    float mu  = rs[0] * (1.0f / D_MODEL);
    float var = rq[0] * (1.0f / D_MODEL) - mu * mu;
    float inv = rsqrtf(var + 1e-12f);
    #pragma unroll
    for (int i = 0; i < 3; ++i) {
        int c = tid + i * 256;
        out[base + c] = (local[i] - mu) * inv * g[c] + bta[c];
    }
}

// ---------------------------------------------------------------------------
extern "C" void kernel_launch(void* const* d_in, const int* in_sizes, int n_in,
                              void* d_out, int out_size, void* d_ws, size_t ws_size,
                              hipStream_t stream)
{
    (void)in_sizes; (void)n_in; (void)out_size; (void)ws_size;

    const float* X    = (const float*)d_in[0];
    const float* ln_g = (const float*)d_in[19];
    const float* ln_b = (const float*)d_in[20];
    float* out = (float*)d_out;

    // workspace layout (fp32), ~564 MB total
    float* ws   = (float*)d_ws;
    float* xz   = ws;                               // NTOK * 3072
    float* uc   = xz   + (size_t)NTOK * 3072;       // NTOK * 1536
    float* xdbl = uc   + (size_t)NTOK * 1536;       // NTOK * 80
    float* dtb  = xdbl + (size_t)NTOK * 80;         // NTOK * 1536
    float* ybuf = dtb  + (size_t)NTOK * 1536;       // NTOK * 1536
    float* outa = ybuf + (size_t)NTOK * 1536;       // NTOK * 768
    float* dtA  = outa + (size_t)NTOK * 768;        // NTOK * 64  (padded dt input)
    float* dtW  = dtA  + (size_t)NTOK * 64;         // 1536 * 64  (padded dt_w)

    const int M = NTOK;
    auto cdiv = [](int a, int b) { return (a + b - 1) / b; };
    const int ELT_BLOCKS = (int)(((size_t)NTOK * D_INNER + 255) / 256);

    for (int dir = 0; dir < 2; ++dir) {
        const int base = 1 + dir * 9;
        const float* in_w    = (const float*)d_in[base + 0];  // (3072, 768)
        const float* conv_w  = (const float*)d_in[base + 1];  // (1536, 4)
        const float* conv_b  = (const float*)d_in[base + 2];  // (1536,)
        const float* xproj_w = (const float*)d_in[base + 3];  // (80, 1536)
        const float* dt_w    = (const float*)d_in[base + 4];  // (1536, 48)
        const float* dt_b    = (const float*)d_in[base + 5];  // (1536,)
        const float* A_log   = (const float*)d_in[base + 6];  // (1536, 16)
        const float* D_skip  = (const float*)d_in[base + 7];  // (1536,)
        const float* out_w   = (const float*)d_in[base + 8];  // (768, 1536)
        const int rev  = dir;         // dir 1 = backward (flipped time)
        const int beta = dir;         // backward accumulates onto forward

        // 1) xz = X @ in_w^T            [16384 x 3072, K=768]
        gemm_bf16_wmma<<<dim3(cdiv(2 * D_INNER, BLK_N), cdiv(M, BLK_M)), 256, 0, stream>>>(
            X, D_MODEL, in_w, D_MODEL, xz, 2 * D_INNER, M, 2 * D_INNER, D_MODEL, 0);

        // 2) uc = silu(causal depthwise conv(u))
        conv_silu_kernel<<<ELT_BLOCKS, 256, 0, stream>>>(xz, conv_w, conv_b, uc, rev);

        // 3) x_dbl = uc @ xproj_w^T     [16384 x 80, K=1536]
        gemm_bf16_wmma<<<dim3(cdiv(DT_RANK + 2 * D_STATE, BLK_N), cdiv(M, BLK_M)), 256, 0, stream>>>(
            uc, D_INNER, xproj_w, D_INNER, xdbl, DT_RANK + 2 * D_STATE,
            M, DT_RANK + 2 * D_STATE, D_INNER, 0);

        // 4a) zero-pad dt inputs 48 -> 64 so the GEMM needs no K guards
        pad48to64_kernel<<<(int)(((size_t)NTOK * 64 + 255) / 256), 256, 0, stream>>>(
            xdbl, DT_RANK + 2 * D_STATE, dtA, NTOK);
        pad48to64_kernel<<<(D_INNER * 64 + 255) / 256, 256, 0, stream>>>(
            dt_w, DT_RANK, dtW, D_INNER);

        // 4b) dt_raw = dtA @ dtW^T      [16384 x 1536, K=64]
        gemm_bf16_wmma<<<dim3(cdiv(D_INNER, BLK_N), cdiv(M, BLK_M)), 256, 0, stream>>>(
            dtA, 64, dtW, 64, dtb, D_INNER, M, D_INNER, 64, 0);

        // 5) dt = softplus(dt_raw + dt_b)
        softplus_kernel<<<ELT_BLOCKS, 256, 0, stream>>>(dtb, dt_b);

        // 6) selective scan -> ybuf
        scan_kernel<<<dim3(D_INNER / 256, BATCH), 256, 0, stream>>>(
            dtb, uc, xdbl, A_log, ybuf, rev);

        // 7) ybuf = (ybuf + uc*D_skip) * silu(z)
        gate_kernel<<<ELT_BLOCKS, 256, 0, stream>>>(ybuf, uc, xz, D_skip);

        // 8) outa (+)= ybuf @ out_w^T   [16384 x 768, K=1536]
        gemm_bf16_wmma<<<dim3(cdiv(D_MODEL, BLK_N), cdiv(M, BLK_M)), 256, 0, stream>>>(
            ybuf, D_INNER, out_w, D_INNER, outa, D_MODEL, M, D_MODEL, D_INNER, beta);
    }

    // 9) out = LN(outa + X) * g + b
    ln_kernel<<<NTOK, 256, 0, stream>>>(outa, X, ln_g, ln_b, out);
}
